// RNN_3461743641041
// MI455X (gfx1250) — matrix operand order repaired and checked by reading, not compiled, not run on hardware
//
#include <hip/hip_runtime.h>
#include <math.h>

// ---------------------------------------------------------------------------
// Elman RNN forward for MI455X (gfx1250, wave32, WMMA + async-to-LDS).
//   T=256, B=128, D_IN=D_H=1024.
//   Phase 1: xi = x @ Wi^T + bi      (one big parallel WMMA GEMM -> d_out)
//   Phase 2: h_t = tanh(xi_t + h @ Wh^T + bh)   (255 sequential WMMA GEMMs)
// f16 operands / f32 accumulation via v_wmma_f32_16x16x32_f16.
// LDS staging is double-buffered; f16 tiles use GLOBAL_LOAD_ASYNC_TO_LDS_B128
// (ASYNCcnt) so next-chunk copies overlap the current chunk's WMMAs.
// ---------------------------------------------------------------------------

typedef _Float16 v16h __attribute__((ext_vector_type(16)));
typedef _Float16 v8h  __attribute__((ext_vector_type(8)));
typedef _Float16 v4h  __attribute__((ext_vector_type(4)));
typedef float    v8f  __attribute__((ext_vector_type(8)));
typedef int      v4i  __attribute__((vector_size(16)));  // matches builtin proto

static constexpr int kT = 256;   // timesteps
static constexpr int kB = 128;   // batch
static constexpr int kD = 1024;  // D_IN == D_H == K == N

#if __has_builtin(__builtin_amdgcn_global_load_async_to_lds_b128)
#define ASYNC_LDS 1
#else
#define ASYNC_LDS 0
#endif

// 16-byte global -> LDS copy: async (direct-to-LDS, ASYNCcnt) when available.
__device__ __forceinline__ void lds_copy16(const _Float16* g, _Float16* l) {
#if ASYNC_LDS
  __builtin_amdgcn_global_load_async_to_lds_b128(
      (__attribute__((address_space(1))) v4i*)(void*)(_Float16*)g,
      (__attribute__((address_space(3))) v4i*)(void*)l,
      /*offset=*/0, /*cpol=*/0);
#else
  *(v8h*)l = *(const v8h*)g;
#endif
}

__device__ __forceinline__ void wait_async() {
#if ASYNC_LDS
#if __has_builtin(__builtin_amdgcn_s_wait_asynccnt)
  __builtin_amdgcn_s_wait_asynccnt(0);
#else
  asm volatile("s_wait_asynccnt 0" ::: "memory");
#endif
#endif
}

// --------------------------- f32 -> f16 convert ----------------------------
__global__ void convert_f32_f16_kernel(const float* __restrict__ src,
                                       _Float16* __restrict__ dst, int n4) {
  int i = blockIdx.x * blockDim.x + threadIdx.x;
  if (i < n4) {
    float4 v = ((const float4*)src)[i];
    v4h h;
    h[0] = (_Float16)v.x; h[1] = (_Float16)v.y;
    h[2] = (_Float16)v.z; h[3] = (_Float16)v.w;
    ((v4h*)dst)[i] = h;
  }
}

// --------------------------- h0 = tanh(xi[0]) ------------------------------
__global__ void tanh_h0_kernel(float* __restrict__ io,
                               _Float16* __restrict__ h16, int n) {
  int i = blockIdx.x * blockDim.x + threadIdx.x;
  if (i < n) {
    float v = tanhf(io[i]);
    io[i] = v;
    h16[i] = (_Float16)v;
  }
}

// ---------------------------------------------------------------------------
// WMMA GEMM:  C[m,n] = sum_k A[m,k] * Bh[n,k]  (+bias[n]) (+carry) (tanh)
//   Bh is f16 [N][K] row-major == B^T, so K is contiguous for both tiles.
//   Wave grid: WM x WN waves; each wave owns TM stacked 16x16 tiles at one
//   16-wide N column.  BM = WM*TM*16, BN = WN*16, BK = 64 (2 WMMA k-steps).
// Fragment layouts per CDNA5 ISA 7.12.2 (wave32):
//   A 16x32: lane {g=l>>4,m=l&15}; halves e<8 -> K=e+8g, e>=8 -> K=e+8+8g
//   B 32x16: lane {g,n=l&15};      halves e<8 -> K=e+8g, e>=8 -> K=e+8+8g
//   C 16x16: elem r, lane l        -> M = r + 8*(l>>4), N = l&15
// ---------------------------------------------------------------------------
template <bool AF16, bool CARRY, bool TANH, bool WRITEH, int WM, int WN, int TM>
__global__ __launch_bounds__(WM* WN * 32) void wmma_gemm_kernel(
    const void* __restrict__ Aptr,        // f32 [M][K] or f16 [M][K]
    const _Float16* __restrict__ Bh,      // f16 [N][K]  (== B^T)
    const float* __restrict__ bias,       // [N]
    float* __restrict__ Cout,             // f32 [M][N]; carry-in if CARRY
    _Float16* __restrict__ Hout) {        // f16 [M][N] if WRITEH
  constexpr int BM  = WM * TM * 16;
  constexpr int BN  = WN * 16;
  constexpr int BK  = 64;
  constexpr int NT  = WM * WN * 32;
  constexpr int PAD = 8;  // rows of 72 halves (144 B): conflict-free b128 LDS

  __shared__ _Float16 As[2][BM][BK + PAD];
  __shared__ _Float16 Bs[2][BN][BK + PAD];

  const int tid  = threadIdx.x;
  const int wave = tid >> 5;   // wave32
  const int lane = tid & 31;
  const int g    = lane >> 4;
  const int l16  = lane & 15;
  const int wn   = wave % WN;
  const int wm   = wave / WN;

  const int block_n = blockIdx.x * BN;
  const int block_m = blockIdx.y * BM;

  // Stage one BK-wide K-chunk into LDS buffer `sb` (8 halves / slot).
  auto stage = [&](int sb, int kc) {
    for (int idx = tid; idx < BM * 8; idx += NT) {
      const int row = idx >> 3;
      const int kk  = (idx & 7) * 8;
      if (AF16) {
        lds_copy16((const _Float16*)Aptr + (size_t)(block_m + row) * kD + kc + kk,
                   &As[sb][row][kk]);
      } else {
        const float* s =
            (const float*)Aptr + (size_t)(block_m + row) * kD + kc + kk;
        float4 f0 = *(const float4*)(s);
        float4 f1 = *(const float4*)(s + 4);
        v8h h;
        h[0] = (_Float16)f0.x; h[1] = (_Float16)f0.y;
        h[2] = (_Float16)f0.z; h[3] = (_Float16)f0.w;
        h[4] = (_Float16)f1.x; h[5] = (_Float16)f1.y;
        h[6] = (_Float16)f1.z; h[7] = (_Float16)f1.w;
        *(v8h*)&As[sb][row][kk] = h;
      }
    }
    for (int idx = tid; idx < BN * 8; idx += NT) {
      const int row = idx >> 3;
      const int kk  = (idx & 7) * 8;
      lds_copy16(Bh + (size_t)(block_n + row) * kD + kc + kk,
                 &Bs[sb][row][kk]);
    }
  };

  v8f acc[TM];
#pragma unroll
  for (int t = 0; t < TM; ++t) acc[t] = (v8f){};

  stage(0, 0);
  wait_async();
  __syncthreads();

  int buf = 0;
  for (int kc = 0; kc < kD; kc += BK) {
    // Kick off next chunk's copies before consuming the current one.
    if (kc + BK < kD) stage(buf ^ 1, kc + BK);

#pragma unroll
    for (int ks = 0; ks < BK; ks += 32) {
      // B fragment: e<8 -> K=ks+e+8g ; e>=8 -> K=ks+e+8+8g
      const v8h b_lo = *(const v8h*)&Bs[buf][wn * 16 + l16][ks + 8 * g];
      const v8h b_hi = *(const v8h*)&Bs[buf][wn * 16 + l16][ks + 16 + 8 * g];
      const v16h bf  = __builtin_shufflevector(b_lo, b_hi, 0, 1, 2, 3, 4, 5, 6,
                                               7, 8, 9, 10, 11, 12, 13, 14, 15);
#pragma unroll
      for (int t = 0; t < TM; ++t) {
        const int arow = (wm * TM + t) * 16 + l16;
        // A fragment: e<8 -> K=ks+e+8g ; e>=8 -> K=ks+e+8+8g
        const v8h a_lo = *(const v8h*)&As[buf][arow][ks + 8 * g];
        const v8h a_hi = *(const v8h*)&As[buf][arow][ks + 16 + 8 * g];
        const v16h af =
            __builtin_shufflevector(a_lo, a_hi, 0, 1, 2, 3, 4, 5, 6, 7, 8, 9,
                                    10, 11, 12, 13, 14, 15);
        acc[t] = __builtin_amdgcn_wmma_f32_16x16x32_f16(
            /*neg_a=*/false, af, /*neg_b=*/false, bf,
            /*c_mod=*/(short)0, acc[t], /*reuse_a=*/false, /*reuse_b=*/false);
      }
    }

    wait_async();     // next chunk's async copies landed in LDS
    __syncthreads();  // ... and everyone is done reading the current buffer
    buf ^= 1;
  }

  // ---- epilogue: C elem r -> row = r + 8g, col = l16 ----
  const int   gcol = block_n + wn * 16 + l16;
  const float bv   = bias[gcol];
#pragma unroll
  for (int t = 0; t < TM; ++t) {
    const int gm0 = block_m + (wm * TM + t) * 16 + 8 * g;
#pragma unroll
    for (int r = 0; r < 8; ++r) {
      const size_t off = (size_t)(gm0 + r) * kD + gcol;
      float v = acc[t][r] + bv;
      if (CARRY) v += Cout[off];
      if (TANH) v = tanhf(v);
      Cout[off] = v;
      if (WRITEH) Hout[off] = (_Float16)v;
    }
  }
}

// ---------------------------------------------------------------------------
extern "C" void kernel_launch(void* const* d_in, const int* in_sizes, int n_in,
                              void* d_out, int out_size, void* d_ws,
                              size_t ws_size, hipStream_t stream) {
  const float* x  = (const float*)d_in[0];  // [T,B,D]
  const float* Wi = (const float*)d_in[1];  // [D,D]
  const float* bi = (const float*)d_in[2];  // [D]
  const float* Wh = (const float*)d_in[3];  // [D,D]
  const float* bh = (const float*)d_in[4];  // [D]
  float* out = (float*)d_out;               // [T,B,D]

  // workspace: Wi_h(2MB) | Wh_h(2MB) | h_a(256KB) | pad(256KB) | h_b(256KB)  = 4.75 MB
  char* ws = (char*)d_ws;
  _Float16* Wi_h = (_Float16*)ws;
  _Float16* Wh_h = (_Float16*)(ws + (2u << 20));
  _Float16* h_a  = (_Float16*)(ws + (4u << 20));
  _Float16* h_b  = (_Float16*)(ws + (4u << 20) + (512u << 10));

  // 1) weights -> f16 (no transpose needed: W[n][k] is already B^T)
  {
    const int n4 = (kD * kD) / 4;
    convert_f32_f16_kernel<<<n4 / 256, 256, 0, stream>>>(Wi, Wi_h, n4);
    convert_f32_f16_kernel<<<n4 / 256, 256, 0, stream>>>(Wh, Wh_h, n4);
  }

  // 2) xi = x @ Wi^T + bi for all T  ->  d_out   (M = T*B = 32768)
  {
    dim3 grid(kD / 64, (kT * kB) / 64);  // 16 x 512 blocks, 8 waves each
    wmma_gemm_kernel<false, false, false, false, 2, 4, 2>
        <<<grid, 256, 0, stream>>>(x, Wi_h, bi, out, nullptr);
  }

  // 3) h0 = tanh(xi[0])  (also emit f16 h into ping buffer)
  {
    const int n = kB * kD;
    tanh_h0_kernel<<<n / 256, 256, 0, stream>>>(out, h_a, n);
  }

  // 4) sequential scan: out[t] = tanh(out[t] + h @ Wh^T + bh)
  for (int t = 1; t < kT; ++t) {
    const _Float16* hp = (t & 1) ? h_a : h_b;  // ping-pong, deterministic
    _Float16*       hn = (t & 1) ? h_b : h_a;
    dim3 grid(kD / 64, kB / 16);  // 16 x 8 = 128 blocks, 4 waves each
    wmma_gemm_kernel<true, true, true, true, 1, 4, 1>
        <<<grid, 128, 0, stream>>>(hp, Wh_h, bh, out + (size_t)t * kB * kD, hn);
  }
}
